// MultiHeadSelfAttention_87316685128022
// MI455X (gfx1250) — matrix operand order repaired and checked by reading, not compiled, not run on hardware
//
#include <hip/hip_runtime.h>
#include <hip/hip_bf16.h>
#include <cstdint>
#include <cstddef>

// ---------------------------------------------------------------------------
// MI455X (gfx1250) multi-head self-attention.
// Compute-bound (~103 GFLOP vs ~50MB HBM traffic) => all matmuls routed
// through v_wmma_f32_16x16x32_bf16 (f32 accum). wave32, 4 waves per block.
// Flash kernel stages K/V tiles into LDS with global_load_async_to_lds_b128
// (double buffered, s_wait_asynccnt) so the DMA overlaps the WMMA pipeline.
// ---------------------------------------------------------------------------

typedef __bf16 bf16;
typedef __attribute__((ext_vector_type(16))) __bf16 v16bf;
typedef __attribute__((ext_vector_type(8)))  float  v8f;
typedef __attribute__((ext_vector_type(4)))  unsigned int u32x4;

#define DEV __device__ __forceinline__

static constexpr int HEADS  = 16;
static constexpr int HDIM   = 64;
static constexpr int DMODEL = 1024;

// ---- bf16 helpers (RNE convert) -------------------------------------------
DEV unsigned short f2bf_bits(float f) {
  unsigned u = __builtin_bit_cast(unsigned, f);
  unsigned r = u + 0x7FFFu + ((u >> 16) & 1u);
  return (unsigned short)(r >> 16);
}
DEV bf16 f2bf(float f) { return __builtin_bit_cast(bf16, f2bf_bits(f)); }
DEV float bf2f(bf16 h) {
  unsigned u = ((unsigned)__builtin_bit_cast(unsigned short, h)) << 16;
  return __builtin_bit_cast(float, u);
}

// ---- WMMA fragment loaders -------------------------------------------------
// A-matrix 16x32 bf16 (ISA 7.12.2): lane&15 = row M, kb=(lane>>4)*8,
//   elems[0..7]=K(kb..kb+7), elems[8..15]=K(kb+16..kb+23).
// B-matrix 32x16: lane&15 = col N, kb=(lane>>4)*8, same K order as A (kb..kb+7, kb+16..kb+23)
//   -> two 16B loads, 32B apart, per lane when B is stored N-major / K-contiguous.
DEV u32x4 ld16B(const bf16* p) { return *(const u32x4*)p; }
DEV v16bf make_frag(u32x4 lo, u32x4 hi) {
  union { u32x4 u[2]; v16bf v; } x;
  x.u[0] = lo; x.u[1] = hi;
  return x.v;
}
DEV v16bf load_afrag(const bf16* row, int kb) {
  return make_frag(ld16B(row + kb), ld16B(row + kb + 16));
}
DEV v16bf load_bfrag(const bf16* row, int kb) {
  return make_frag(ld16B(row + kb), ld16B(row + kb + 16));
}
DEV v8f wmma_bf16(v16bf a, v16bf b, v8f c) {
  return __builtin_amdgcn_wmma_f32_16x16x32_bf16(
      false, a, false, b, (short)0, c, false, false);
}

// ---- row-max reduction over each 16-lane half, pure VALU DPP --------------
// quad_perm xor1 (0xB1), quad_perm xor2 (0x4E), row_ror:4 (0x124),
// row_ror:8 (0x128). DPP "row" = 16 lanes -> exactly one wave32 half.
template <int CTRL>
DEV float fmax_dpp(float v) {
  int t = __builtin_amdgcn_update_dpp(0, __builtin_bit_cast(int, v), CTRL,
                                      0xf, 0xf, true);
  return fmaxf(v, __builtin_bit_cast(float, t));
}
DEV float rowmax16(float v) {
  v = fmax_dpp<0xB1>(v);
  v = fmax_dpp<0x4E>(v);
  v = fmax_dpp<0x124>(v);
  v = fmax_dpp<0x128>(v);
  return v;
}

// ---- CDNA5 async LDS staging ----------------------------------------------
DEV void async_copy_b128(unsigned lds_addr, const void* gaddr) {
  asm volatile("global_load_async_to_lds_b128 %0, %1, off"
               :: "v"(lds_addr), "v"((unsigned long long)(uintptr_t)gaddr)
               : "memory");
}
DEV void wait_async0() {
  asm volatile("s_wait_asynccnt 0x0" ::: "memory");
}

// ===========================================================================
// Kernel 1: f32 -> bf16 convert
// ===========================================================================
__global__ void cvt_f32_bf16(const float* __restrict__ in,
                             bf16* __restrict__ out, int n) {
  int i = blockIdx.x * blockDim.x + threadIdx.x;
  if (i < n) out[i] = f2bf(in[i]);
}

// ===========================================================================
// Kernel 2: C[M,N] = A[M,K] * W[N,K]^T  (bf16 in, f32 accum)
// wave: 32(M) x 64(N); block = 4 waves -> 128(M) x 64(N).
// ===========================================================================
DEV void storeC(float* p, float v) { *p = v; }
DEV void storeC(bf16* p, float v)  { *p = f2bf(v); }

template <typename OutT>
__global__ __launch_bounds__(128) void gemm_bf16_bt(
    const bf16* __restrict__ A, const bf16* __restrict__ W,
    OutT* __restrict__ C, int M, int N, int K) {
  const int lane = threadIdx.x & 31;
  const int wave = threadIdx.x >> 5;
  const int half = lane >> 4;
  const int l15  = lane & 15;
  const int nBlk = N / 64;
  const int m0 = (blockIdx.x / nBlk) * 128 + wave * 32;
  const int n0 = (blockIdx.x % nBlk) * 64;
  const int kbA = half * 8;
  const int kbB = half * 8;

  v8f acc[2][4];
#pragma unroll
  for (int mt = 0; mt < 2; ++mt)
#pragma unroll
    for (int nt = 0; nt < 4; ++nt) acc[mt][nt] = (v8f)(0.0f);

  const bf16* arow0 = A + (size_t)(m0 + l15) * K;
  const bf16* arow1 = A + (size_t)(m0 + 16 + l15) * K;
  const bf16* brow[4];
#pragma unroll
  for (int nt = 0; nt < 4; ++nt)
    brow[nt] = W + (size_t)(n0 + nt * 16 + l15) * K;

  for (int k0 = 0; k0 < K; k0 += 32) {
    __builtin_prefetch(arow0 + k0 + 256, 0, 0);  // global_prefetch_b8
    v16bf a0 = load_afrag(arow0 + k0, kbA);
    v16bf a1 = load_afrag(arow1 + k0, kbA);
#pragma unroll
    for (int nt = 0; nt < 4; ++nt) {
      v16bf b = load_bfrag(brow[nt] + k0, kbB);
      acc[0][nt] = wmma_bf16(a0, b, acc[0][nt]);
      acc[1][nt] = wmma_bf16(a1, b, acc[1][nt]);
    }
  }

#pragma unroll
  for (int mt = 0; mt < 2; ++mt)
#pragma unroll
    for (int nt = 0; nt < 4; ++nt)
#pragma unroll
      for (int j = 0; j < 8; ++j) {
        int row = m0 + mt * 16 + half * 8 + j;
        int col = n0 + nt * 16 + l15;
        storeC(C + (size_t)row * N + col, acc[mt][nt][j]);
      }
}

// ===========================================================================
// Kernel 3: RoPE + reshape q,k:  [b*t, 1024] -> [b,h,t,64]
// ===========================================================================
__global__ void rope_reshape(const bf16* __restrict__ qin,
                             const bf16* __restrict__ kin,
                             bf16* __restrict__ qout,
                             bf16* __restrict__ kout, int T, int total) {
  int gid = blockIdx.x * blockDim.x + threadIdx.x;
  if (gid >= total) return;                 // total = B*T*512 pairs
  int row = gid / 512;                      // b*T + t
  int pr  = gid % 512;
  int t = row % T, b = row / T;
  int h = pr >> 5, i = pr & 31;
  float inv = __expf(-(float)(2 * i) * (9.210340371976184f / 64.0f));
  float ang = (float)t * inv;
  float sn = __sinf(ang), cs = __cosf(ang);
  size_t src = (size_t)row * DMODEL + h * HDIM + 2 * i;
  size_t dst = ((size_t)(b * HEADS + h) * T + t) * HDIM + 2 * i;
  {
    float x1 = bf2f(qin[src]), x2 = bf2f(qin[src + 1]);
    qout[dst]     = f2bf(x1 * cs - x2 * sn);
    qout[dst + 1] = f2bf(x1 * sn + x2 * cs);
  }
  {
    float x1 = bf2f(kin[src]), x2 = bf2f(kin[src + 1]);
    kout[dst]     = f2bf(x1 * cs - x2 * sn);
    kout[dst + 1] = f2bf(x1 * sn + x2 * cs);
  }
}

// ===========================================================================
// Kernel 4: V transpose: [b*t, 1024] -> [b,h,64,t]
// ===========================================================================
__global__ void v_transpose(const bf16* __restrict__ vin,
                            bf16* __restrict__ vT, int T, int total) {
  int gid = blockIdx.x * blockDim.x + threadIdx.x;
  if (gid >= total) return;                 // total = B*T*1024
  int row = gid / DMODEL, e = gid % DMODEL;
  int t = row % T, b = row / T;
  int h = e >> 6, d = e & 63;
  vT[((size_t)(b * HEADS + h) * HDIM + d) * T + t] = vin[gid];
}

// ===========================================================================
// Kernel 5: causal flash attention. Block = (b, h, 64-row q block), 4 waves.
// K (32x64) / V (64x32) tiles async-DMA'd into double-buffered LDS once per
// block; each wave owns 16 q rows; l carried as a ones-column WMMA
// accumulator; row max via DPP; P staged through LDS for the PV A-fragment.
// ===========================================================================
__global__ __launch_bounds__(128) void flash_attn(
    const bf16* __restrict__ Q, const bf16* __restrict__ Kt,
    const bf16* __restrict__ Vt, bf16* __restrict__ attn, int T) {
  const int tid  = threadIdx.x;
  const int lane = tid & 31;
  const int wave = tid >> 5;
  const int half = lane >> 4;
  const int l15  = lane & 15;
  const int kbA  = half * 8;
  const int kbB  = half * 8;

  const int qBlocks = T / 64;
  const int qblk = blockIdx.x % qBlocks;
  const int bh   = blockIdx.x / qBlocks;    // b*HEADS + h
  const int b = bh / HEADS, h = bh % HEADS;
  const int qb = qblk * 64 + wave * 16;

  __shared__ __align__(16) bf16 Klds[2][32 * HDIM];   // 2 x 4KB
  __shared__ __align__(16) bf16 Vlds[2][HDIM * 32];   // 2 x 4KB
  __shared__ __align__(16) bf16 Pst[4][16 * 32];      // 4KB

  const bf16* Qp = Q  + (size_t)bh * T * HDIM;
  const bf16* Kp = Kt + (size_t)bh * T * HDIM;
  const bf16* Vp = Vt + (size_t)bh * HDIM * T;

  // Q A-fragments for both hd K-steps (fixed for the whole kv loop)
  const bf16* qrow = Qp + (size_t)(qb + l15) * HDIM;
  v16bf aq0 = load_afrag(qrow +  0, kbA);
  v16bf aq1 = load_afrag(qrow + 32, kbA);

  // all-ones B fragment: l obeys the same recurrence as O
  v16bf ones;
#pragma unroll
  for (int i = 0; i < 16; ++i) ones[i] = f2bf(1.0f);

  float mrow[8];
#pragma unroll
  for (int j = 0; j < 8; ++j) mrow[j] = -3.0e38f;
  v8f o[4], accL = (v8f)(0.0f);
#pragma unroll
  for (int nt = 0; nt < 4; ++nt) o[nt] = (v8f)(0.0f);

  // ---- async stage of one kv tile into LDS buffer `buf` ----
  auto stage = [&](int kv0, int buf) {
    // K: 32 rows x 64 elems, contiguous 4KB -> 256 x 16B chunks
    unsigned kb_ = (unsigned)(uintptr_t)&Klds[buf][0];
    const char* kg = (const char*)(Kp + (size_t)kv0 * HDIM);
#pragma unroll
    for (int r = 0; r < 2; ++r) {
      int c = tid + r * 128;
      async_copy_b128(kb_ + c * 16, kg + c * 16);
    }
    // V: 64 rows x 32 elems; row r at Vp + r*T + kv0 (64B = 4 chunks)
    unsigned vb_ = (unsigned)(uintptr_t)&Vlds[buf][0];
#pragma unroll
    for (int r = 0; r < 2; ++r) {
      int c = tid + r * 128;
      const char* vg =
          (const char*)(Vp + (size_t)(c >> 2) * T + kv0) + (c & 3) * 16;
      async_copy_b128(vb_ + c * 16, vg);
    }
  };

  const int nIter = (qblk * 64 + 64) / 32;  // causal bound, block-uniform
  const int nFull = (qblk * 64) / 32;       // iterations needing no mask
  constexpr float SC = 0.125f * 1.4426950408889634f;  // 1/sqrt(64) * log2(e)

  stage(0, 0);

  for (int it = 0; it < nIter; ++it) {
    const int kv0 = it * 32;
    const int buf = it & 1;
    wait_async0();
    __syncthreads();              // tile `buf` resident; prior reads retired
    if (it + 1 < nIter) stage(kv0 + 32, buf ^ 1);

    // ---- S = Q K^T, two 16-col subtiles, K from LDS ----
    v8f s[2];
#pragma unroll
    for (int c = 0; c < 2; ++c) {
      const bf16* krow = &Klds[buf][(c * 16 + l15) * HDIM];
      v16bf bk0 = load_bfrag(krow +  0, kbB);
      v16bf bk1 = load_bfrag(krow + 32, kbB);
      v8f t = (v8f)(0.0f);
      t = wmma_bf16(aq0, bk0, t);
      t = wmma_bf16(aq1, bk1, t);
      s[c] = t;
    }

    // ---- online softmax (log2 domain); mask only on 2 diagonal tiles ----
    float mnew[8], sc[8];
    if (it < nFull) {
#pragma unroll
      for (int j = 0; j < 8; ++j) {
        float s0 = s[0][j] * SC, s1 = s[1][j] * SC;
        s[0][j] = s0; s[1][j] = s1;
        mnew[j] = fmaxf(mrow[j], rowmax16(fmaxf(s0, s1)));
      }
    } else {
      int col0 = kv0 + l15, col1 = kv0 + 16 + l15;
#pragma unroll
      for (int j = 0; j < 8; ++j) {
        int qr_ = qb + half * 8 + j;
        float s0 = (col0 <= qr_) ? s[0][j] * SC : -1.0e9f;
        float s1 = (col1 <= qr_) ? s[1][j] * SC : -1.0e9f;
        s[0][j] = s0; s[1][j] = s1;
        mnew[j] = fmaxf(mrow[j], rowmax16(fmaxf(s0, s1)));
      }
    }
#pragma unroll
    for (int j = 0; j < 8; ++j) {
      sc[j] = exp2f(mrow[j] - mnew[j]);
      mrow[j] = mnew[j];
      float p0 = exp2f(s[0][j] - mnew[j]);
      float p1 = exp2f(s[1][j] - mnew[j]);
      int prow = half * 8 + j;
      Pst[wave][prow * 32 + l15]      = f2bf(p0);
      Pst[wave][prow * 32 + 16 + l15] = f2bf(p1);
    }
#pragma unroll
    for (int nt = 0; nt < 4; ++nt)
#pragma unroll
      for (int j = 0; j < 8; ++j) o[nt][j] *= sc[j];
#pragma unroll
    for (int j = 0; j < 8; ++j) accL[j] *= sc[j];

    __syncthreads();              // P visible (uniform kv range per block)

    // ---- P (A-frag from LDS) x V rows (LDS) + ones column for l ----
    union { u32x4 u[2]; v16bf v; } pu;
    pu.u[0] = *(const u32x4*)&Pst[wave][l15 * 32 + kbA];
    pu.u[1] = *(const u32x4*)&Pst[wave][l15 * 32 + kbA + 16];
    v16bf ap = pu.v;
#pragma unroll
    for (int nt = 0; nt < 4; ++nt) {
      const bf16* vrow = &Vlds[buf][(nt * 16 + l15) * 32];
      v16bf bv = load_bfrag(vrow, kbB);
      o[nt] = wmma_bf16(ap, bv, o[nt]);
    }
    accL = wmma_bf16(ap, ones, accL);
  }

  // ---- epilogue: O / l -> attn[b,t, h*64 + d] ----
  float linv[8];
#pragma unroll
  for (int j = 0; j < 8; ++j) linv[j] = 1.0f / accL[j];
#pragma unroll
  for (int nt = 0; nt < 4; ++nt)
#pragma unroll
    for (int j = 0; j < 8; ++j) {
      int row = qb + half * 8 + j;
      int col = h * HDIM + nt * 16 + l15;
      attn[(size_t)(b * T + row) * DMODEL + col] = f2bf(o[nt][j] * linv[j]);
    }
}

// ===========================================================================
// Launcher
// ===========================================================================
extern "C" void kernel_launch(void* const* d_in, const int* in_sizes, int n_in,
                              void* d_out, int out_size, void* d_ws,
                              size_t ws_size, hipStream_t stream) {
  const float* x  = (const float*)d_in[0];
  const float* wq = (const float*)d_in[1];
  const float* wk = (const float*)d_in[2];
  const float* wv = (const float*)d_in[3];
  const float* wo = (const float*)d_in[4];

  const int D = DMODEL;
  const int B = 2;
  const int T = in_sizes[0] / (B * D);      // 2048
  const int M = B * T;                      // 4096 rows

  char* ws = (char*)d_ws;
  size_t off = 0;
  auto carve = [&](size_t bytes) {
    char* p = ws + off;
    off += (bytes + 255) & ~(size_t)255;
    return p;
  };
  const size_t actBytes = (size_t)M * D * sizeof(bf16);     // 8 MB
  const size_t wBytes   = (size_t)D * D * sizeof(bf16);     // 2 MB
  bf16* xb   = (bf16*)carve(actBytes);
  bf16* wqb  = (bf16*)carve(wBytes);
  bf16* wkb  = (bf16*)carve(wBytes);
  bf16* wvb  = (bf16*)carve(wBytes);
  bf16* wob  = (bf16*)carve(wBytes);
  bf16* qbuf = (bf16*)carve(actBytes);
  bf16* kbuf = (bf16*)carve(actBytes);
  bf16* vbuf = (bf16*)carve(actBytes);
  bf16* qr   = (bf16*)carve(actBytes);
  bf16* kr   = (bf16*)carve(actBytes);
  bf16* vT   = (bf16*)carve(actBytes);
  bf16* attn = (bf16*)carve(actBytes);
  (void)ws_size;

  {
    int n = M * D;
    cvt_f32_bf16<<<(n + 255) / 256, 256, 0, stream>>>(x, xb, n);
    int nw = D * D;
    cvt_f32_bf16<<<(nw + 255) / 256, 256, 0, stream>>>(wq, wqb, nw);
    cvt_f32_bf16<<<(nw + 255) / 256, 256, 0, stream>>>(wk, wkb, nw);
    cvt_f32_bf16<<<(nw + 255) / 256, 256, 0, stream>>>(wv, wvb, nw);
    cvt_f32_bf16<<<(nw + 255) / 256, 256, 0, stream>>>(wo, wob, nw);
  }

  {
    dim3 grid((M / 128) * (D / 64));
    gemm_bf16_bt<bf16><<<grid, 128, 0, stream>>>(xb, wqb, qbuf, M, D, D);
    gemm_bf16_bt<bf16><<<grid, 128, 0, stream>>>(xb, wkb, kbuf, M, D, D);
    gemm_bf16_bt<bf16><<<grid, 128, 0, stream>>>(xb, wvb, vbuf, M, D, D);
  }

  {
    int pairs = B * T * (D / 2);
    rope_reshape<<<(pairs + 255) / 256, 256, 0, stream>>>(qbuf, kbuf, qr, kr,
                                                          T, pairs);
    int n = M * D;
    v_transpose<<<(n + 255) / 256, 256, 0, stream>>>(vbuf, vT, T, n);
  }

  {
    dim3 grid(B * HEADS * (T / 64));
    flash_attn<<<grid, 128, 0, stream>>>(qr, kr, vT, attn, T);
  }

  {
    dim3 grid((M / 128) * (D / 64));
    gemm_bf16_bt<float><<<grid, 128, 0, stream>>>(attn, wob, (float*)d_out, M,
                                                  D, D);
  }
}
